// Gumbel_MoE_DSFNet_11605001634409
// MI455X (gfx1250) — matrix operand order repaired and checked by reading, not compiled, not run on hardware
//
#include <hip/hip_runtime.h>
#include <hip/hip_bf16.h>
#include <stdint.h>

// ---------------------------------------------------------------------------
// Types for CDNA5 WMMA (gfx1250): v_wmma_f32_16x16x32_bf16
// ---------------------------------------------------------------------------
typedef __attribute__((ext_vector_type(16))) __bf16 v16bf;
typedef __attribute__((ext_vector_type(8)))  __bf16 v8bf;
typedef __attribute__((ext_vector_type(8)))  float  v8f;

#define USE_ASYNC_LDS 1

__device__ __forceinline__ __bf16 f2bf(float f) {
    union { float f; unsigned u; } v; v.f = f;
    unsigned r = v.u + 0x7FFFu + ((v.u >> 16) & 1u);   // round-to-nearest-even
    union { unsigned short s; __bf16 b; } o; o.s = (unsigned short)(r >> 16);
    return o.b;
}
__device__ __forceinline__ float bf2f(__bf16 b) {
    union { unsigned short s; __bf16 b; } i; i.b = b;
    union { unsigned u; float f; } o; o.u = ((unsigned)i.s) << 16;
    return o.f;
}
__device__ __forceinline__ __bf16 to_bf(float f)  { return f2bf(f); }
__device__ __forceinline__ __bf16 to_bf(__bf16 b) { return b; }
__device__ __forceinline__ void store_out(float*  p, float v) { *p = v; }
__device__ __forceinline__ void store_out(__bf16* p, float v) { *p = f2bf(v); }

// Async global -> LDS copy, 16B (GLOBAL_LOAD_ASYNC_TO_LDS_B128, ASYNCcnt).
// Generic shared pointer: low 32 bits == LDS byte address (flat aperture rule).
__device__ __forceinline__ void async_copy_b128(void* lds_dst, const void* gsrc) {
#if USE_ASYNC_LDS
    unsigned loff = (unsigned)(uintptr_t)lds_dst;
    asm volatile("global_load_async_to_lds_b128 %0, %1, off"
                 :: "v"(loff), "v"(gsrc) : "memory");
#else
    *reinterpret_cast<float4*>(lds_dst) = *reinterpret_cast<const float4*>(gsrc);
#endif
}
__device__ __forceinline__ void wait_async0() {
#if USE_ASYNC_LDS
#if defined(__has_builtin) && __has_builtin(__builtin_amdgcn_s_wait_asynccnt)
    __builtin_amdgcn_s_wait_asynccnt(0);
#else
    asm volatile("s_wait_asynccnt 0x0" ::: "memory");
#endif
#endif
}

// ---------------------------------------------------------------------------
// Weight pack: f32 [E][M][K] -> bf16 [E][Mpad][Kpad], zero padded
// ---------------------------------------------------------------------------
__global__ void pack_w(const float* __restrict__ src, __bf16* __restrict__ dst,
                       int E, int M, int K, int Mpad, int Kpad) {
    int idx = blockIdx.x * 256 + threadIdx.x;
    int total = E * Mpad * Kpad;
    if (idx >= total) return;
    int e = idx / (Mpad * Kpad);
    int r = idx - e * Mpad * Kpad;
    int m = r / Kpad;
    int k = r - m * Kpad;
    float v = (m < M && k < K) ? src[((size_t)e * M + m) * K + k] : 0.0f;
    dst[idx] = f2bf(v);
}

// ---------------------------------------------------------------------------
// Implicit-GEMM conv via WMMA (bf16 in, f32 accumulate).
//   D[m][n] = sum_k W[m][k] * im2col(in)[k][n]   (+ bias, optional ReLU)
// Block: 256 threads = 8 wave32 arranged 4(M) x 2(N); block tile 64(M) x 64(N).
// Each wave owns two 16x16 accumulators; K consumed in 64-chunks through LDS
// => 4 v_wmma per barrier pair per wave.
// Weights staged with GLOBAL_LOAD_ASYNC_TO_LDS_B128; im2col patches built
// in-VALU (bf16 convert + zero pad). Per-batch MoE routing via route[b].
// ---------------------------------------------------------------------------
template <typename TIN, typename TOUT, int KH, int KW>
__global__ __launch_bounds__(256) void conv_wmma(
    const TIN* __restrict__ in, const __bf16* __restrict__ wts,
    const float* __restrict__ bias, const int* __restrict__ route,
    TOUT* __restrict__ out,
    int Cin, int H, int W, int stride, int pad,
    int Hout, int Wout, int Mreal, int Mpad, int Kreal, int Kpad, int relu)
{
    __shared__ __align__(16) __bf16 As[64 * 64];  // A tile: [m][k]
    __shared__ __align__(16) __bf16 Bs[64 * 64];  // B tile: [n][k] (k contiguous)
    constexpr int KHW = KH * KW;

    const int tid  = threadIdx.x;
    const int lane = tid & 31;
    const int wave = tid >> 5;
    const int b     = blockIdx.z;
    const int mtile = blockIdx.y * 64;
    const int ntile = blockIdx.x * 64;
    const int Nout  = Hout * Wout;
    const int e = route ? route[b] : 0;
    const __bf16* Wp = wts + (size_t)e * Mpad * Kpad;
    const TIN* inb = in + (size_t)b * Cin * H * W;

    const int wm  = (wave >> 1) * 16;   // wave M offset in block tile
    const int wn0 = (wave & 1) * 32;    // wave N offset (two 16-wide tiles)

    // B loader: thread -> (n, 16 consecutive k)
    const int bn   = tid >> 2;          // 0..63
    const int bks  = (tid & 3) * 16;    // 0,16,32,48
    const int nIdx = ntile + bn;
    const int oy = nIdx / Wout, ox = nIdx - oy * Wout;
    const int iy0 = oy * stride - pad, ix0 = ox * stride - pad;

    // A loader: thread -> (row, 16 consecutive k) == two async b128
    const int ar = tid >> 2;            // 0..63
    const int ak = (tid & 3) * 16;      // 0,16,32,48 (halves)

    v8f acc0 = {}, acc1 = {};

    for (int k0 = 0; k0 < Kpad; k0 += 64) {
        // ---- stage A (weights) 64x64 bf16 into LDS via async DMA ----
        {
            int gm = mtile + ar;
            if (gm < Mpad) {
                const __bf16* src = Wp + (size_t)gm * Kpad + k0 + ak;
                async_copy_b128(&As[ar * 64 + ak],     src);
                async_copy_b128(&As[ar * 64 + ak + 8], src + 8);
            } else {
                float4 z{};
                *reinterpret_cast<float4*>(&As[ar * 64 + ak])     = z;
                *reinterpret_cast<float4*>(&As[ar * 64 + ak + 8]) = z;
            }
        }
        // ---- stage B (im2col patches) 64n x 64k into LDS ----
        #pragma unroll
        for (int j = 0; j < 16; ++j) {
            int kk = k0 + bks + j;
            __bf16 val = f2bf(0.0f);
            if (kk < Kreal) {
                int ci  = kk / KHW;           // compile-time divisor
                int rem = kk - ci * KHW;
                int ky  = rem / KW;
                int kx  = rem - ky * KW;
                int iy  = iy0 + ky;
                int ix  = ix0 + kx;
                if (iy >= 0 && iy < H && ix >= 0 && ix < W)
                    val = to_bf(inb[((size_t)ci * H + iy) * W + ix]);
            }
            Bs[bn * 64 + bks + j] = val;
        }
        wait_async0();
        __syncthreads();

        // ---- 2 K-subchunks x 2 N-tiles = 4 WMMAs (k = 16*hi + i in both A and B) ----
        const int hi = lane >> 4;
        const int am = wm + (lane & 15);
        const int bc0 = wn0 + (lane & 15);
        #pragma unroll
        for (int s = 0; s < 2; ++s) {
            v8bf a0 = *reinterpret_cast<const v8bf*>(&As[am * 64 + s * 32 + hi * 16]);
            v8bf a1 = *reinterpret_cast<const v8bf*>(&As[am * 64 + s * 32 + hi * 16 + 8]);
            v16bf af = __builtin_shufflevector(a0, a1,
                        0,1,2,3,4,5,6,7,8,9,10,11,12,13,14,15);

            v8bf b0 = *reinterpret_cast<const v8bf*>(&Bs[bc0 * 64 + s * 32 + hi * 16]);
            v8bf b1 = *reinterpret_cast<const v8bf*>(&Bs[bc0 * 64 + s * 32 + hi * 16 + 8]);
            v16bf bf0 = __builtin_shufflevector(b0, b1,
                        0,1,2,3,4,5,6,7,8,9,10,11,12,13,14,15);
            acc0 = __builtin_amdgcn_wmma_f32_16x16x32_bf16(false, af, false, bf0,
                                                           (short)0, acc0, false, false);

            v8bf c0 = *reinterpret_cast<const v8bf*>(&Bs[(bc0 + 16) * 64 + s * 32 + hi * 16]);
            v8bf c1 = *reinterpret_cast<const v8bf*>(&Bs[(bc0 + 16) * 64 + s * 32 + hi * 16 + 8]);
            v16bf bf1 = __builtin_shufflevector(c0, c1,
                        0,1,2,3,4,5,6,7,8,9,10,11,12,13,14,15);
            acc1 = __builtin_amdgcn_wmma_f32_16x16x32_bf16(false, af, false, bf1,
                                                           (short)0, acc1, false, false);
        }
        __syncthreads();
    }

    // ---- epilogue: bias + ReLU + store (C layout: VGPR r -> M=(lane>>4)*8+r) ----
    const float* bp = bias ? (bias + (size_t)e * Mreal) : nullptr;
    #pragma unroll
    for (int r = 0; r < 8; ++r) {
        int m = mtile + wm + (lane >> 4) * 8 + r;
        if (m >= Mreal) continue;
        float bv = bp ? bp[m] : 0.0f;
        #pragma unroll
        for (int t = 0; t < 2; ++t) {
            int n = ntile + wn0 + t * 16 + (lane & 15);
            if (n < Nout) {
                float v = (t == 0 ? acc0[r] : acc1[r]) + bv;
                if (relu) v = v > 0.0f ? v : 0.0f;
                store_out(&out[((size_t)b * Mreal + m) * Nout + n], v);
            }
        }
    }
}

// ---------------------------------------------------------------------------
// Gating: BN + ReLU + 3x3/s2 maxpool (pad 1) + spatial mean -> pooled[b][c]
// ---------------------------------------------------------------------------
__global__ __launch_bounds__(256) void bn_pool_mean(
    const __bf16* __restrict__ f, const float* __restrict__ gamma,
    const float* __restrict__ beta, const float* __restrict__ mu,
    const float* __restrict__ var, float* __restrict__ pooled)
{
    int bc = blockIdx.x;
    int c  = bc & 63;
    float scale = gamma[c] * rsqrtf(var[c] + 1e-5f);
    float shift = beta[c] - mu[c] * scale;
    const __bf16* base = f + (size_t)bc * 256 * 256;

    float sum = 0.0f;
    for (int i = 0; i < 64; ++i) {
        int p  = threadIdx.x + i * 256;
        int py = p >> 7, px = p & 127;
        float mx = -3.0e38f;
        #pragma unroll
        for (int dy = -1; dy <= 1; ++dy) {
            int iy = py * 2 + dy;
            if (iy < 0 || iy > 255) continue;
            #pragma unroll
            for (int dx = -1; dx <= 1; ++dx) {
                int ix = px * 2 + dx;
                if (ix < 0 || ix > 255) continue;
                float v = bf2f(base[iy * 256 + ix]) * scale + shift;
                mx = v > mx ? v : mx;
            }
        }
        mx = mx > 0.0f ? mx : 0.0f;
        sum += mx;
    }
    __shared__ float red[256];
    red[threadIdx.x] = sum;
    __syncthreads();
    for (int s = 128; s > 0; s >>= 1) {
        if (threadIdx.x < s) red[threadIdx.x] += red[threadIdx.x + s];
        __syncthreads();
    }
    if (threadIdx.x == 0) pooled[bc] = red[0] * (1.0f / 16384.0f);
}

// ---------------------------------------------------------------------------
// FC -> top-1 routing + aux (load-balance) loss. One wave32.
// ---------------------------------------------------------------------------
__global__ void route_kernel(const float* __restrict__ pooled,
                             const float* __restrict__ fcw,
                             const float* __restrict__ fcb,
                             int* __restrict__ route, float* __restrict__ aux_out)
{
    __shared__ float probs[8][3];
    __shared__ int   rt[8];
    int b = threadIdx.x;
    if (b < 8) {
        float l[3];
        #pragma unroll
        for (int e = 0; e < 3; ++e) {
            float s = fcb[e];
            for (int c = 0; c < 64; ++c) s += pooled[b * 64 + c] * fcw[e * 64 + c];
            l[e] = s;
        }
        int am = 0;
        if (l[1] > l[0])  am = 1;
        if (l[2] > l[am]) am = 2;            // first max wins (jax top_k tie rule)
        rt[b] = am; route[b] = am;
        float mx = fmaxf(l[0], fmaxf(l[1], l[2]));
        float e0 = __expf(l[0] - mx), e1 = __expf(l[1] - mx), e2 = __expf(l[2] - mx);
        float inv = 1.0f / (e0 + e1 + e2);
        probs[b][0] = e0 * inv; probs[b][1] = e1 * inv; probs[b][2] = e2 * inv;
    }
    __syncthreads();
    if (threadIdx.x == 0) {
        float aux = 0.0f;
        #pragma unroll
        for (int e = 0; e < 3; ++e) {
            float d = 0.0f, pr = 0.0f;
            for (int bb = 0; bb < 8; ++bb) {
                d  += (rt[bb] == e) ? 1.0f : 0.0f;
                pr += probs[bb][e];
            }
            aux += (d * 0.125f) * (pr * 0.125f);
        }
        *aux_out = 0.01f * (aux * 3.0f);     // LOSS_COEF * NUM_EXPERTS * sum
    }
}

// ---------------------------------------------------------------------------
// Scatter head output (b,5,128,128) f32 -> concatenated (hm | wh | reg)
// ---------------------------------------------------------------------------
__global__ void scatter_out(const float* __restrict__ src, float* __restrict__ dst) {
    int idx = blockIdx.x * 256 + threadIdx.x;
    const int TOT = 8 * 5 * 16384;
    if (idx >= TOT) return;
    int b = idx / (5 * 16384);
    int r = idx - b * 5 * 16384;
    int c = r / 16384;
    int p = r - c * 16384;
    float v = src[idx];
    size_t o;
    if (c == 0)      o = (size_t)b * 16384 + p;                            // hm
    else if (c < 3)  o = 131072 + ((size_t)b * 2 + (c - 1)) * 16384 + p;   // wh
    else             o = 393216 + ((size_t)b * 2 + (c - 3)) * 16384 + p;   // reg
    dst[o] = v;
}

// ---------------------------------------------------------------------------
extern "C" void kernel_launch(void* const* d_in, const int* in_sizes, int n_in,
                              void* d_out, int out_size, void* d_ws, size_t ws_size,
                              hipStream_t stream)
{
    (void)in_sizes; (void)n_in; (void)ws_size;
    const float* x       = (const float*)d_in[0];
    const float* g_convw = (const float*)d_in[1];
    const float* g_gamma = (const float*)d_in[2];
    const float* g_beta  = (const float*)d_in[3];
    const float* g_mean  = (const float*)d_in[4];
    const float* g_var   = (const float*)d_in[5];
    const float* g_fcw   = (const float*)d_in[6];
    const float* g_fcb   = (const float*)d_in[7];
    const float* e_c1w   = (const float*)d_in[8];
    const float* e_c2w   = (const float*)d_in[9];
    const float* e_h1w   = (const float*)d_in[10];
    const float* e_h1b   = (const float*)d_in[11];
    const float* e_h2w   = (const float*)d_in[12];
    const float* e_h2b   = (const float*)d_in[13];
    float* out = (float*)d_out;

    char* ws = (char*)d_ws;
    size_t cur = 0;
    auto carve = [&](size_t bytes) {
        char* p = ws + cur;
        cur = (cur + bytes + 255) & ~(size_t)255;
        return p;
    };
    __bf16* wg_g   = (__bf16*)carve((size_t)1 * 64  * 768  * 2);
    __bf16* wg_c1  = (__bf16*)carve((size_t)3 * 64  * 768  * 2);
    __bf16* wg_c2  = (__bf16*)carve((size_t)3 * 128 * 576  * 2);
    __bf16* wg_h1  = (__bf16*)carve((size_t)3 * 128 * 1152 * 2);
    __bf16* wg_h2  = (__bf16*)carve((size_t)3 * 16  * 128  * 2);
    __bf16* buf_a  = (__bf16*)carve((size_t)8 * 64  * 256 * 256 * 2); // gate conv out, then conv1 out
    __bf16* buf_b  = (__bf16*)carve((size_t)8 * 128 * 128 * 128 * 2); // conv2 out
    __bf16* buf_c  = (__bf16*)carve((size_t)8 * 128 * 128 * 128 * 2); // head1 out
    float*  buf_h2 = (float*) carve((size_t)8 * 5   * 128 * 128 * 4); // head2 out
    float*  pooled = (float*) carve(8 * 64 * 4);
    int*    route  = (int*)   carve(8 * 4);

    // 1) pack weights -> bf16 [E][Mpad][Kpad] (Kpad multiple of 64)
    pack_w<<<(1*64*768  + 255)/256, 256, 0, stream>>>(g_convw, wg_g,  1, 64,  735,  64,  768);
    pack_w<<<(3*64*768  + 255)/256, 256, 0, stream>>>(e_c1w,   wg_c1, 3, 64,  735,  64,  768);
    pack_w<<<(3*128*576 + 255)/256, 256, 0, stream>>>(e_c2w,   wg_c2, 3, 128, 576,  128, 576);
    pack_w<<<(3*128*1152+ 255)/256, 256, 0, stream>>>(e_h1w,   wg_h1, 3, 128, 1152, 128, 1152);
    pack_w<<<(3*16*128  + 255)/256, 256, 0, stream>>>(e_h2w,   wg_h2, 3, 5,   128,  16,  128);

    // 2) gating conv (WMMA), raw output bf16
    dim3 g1(65536 / 64, 1, 8);
    conv_wmma<float, __bf16, 7, 7><<<g1, 256, 0, stream>>>(
        x, wg_g, nullptr, nullptr, buf_a,
        15, 512, 512, 2, 3, 256, 256, 64, 64, 735, 768, 0);

    // 3) BN + ReLU + maxpool + mean
    bn_pool_mean<<<8 * 64, 256, 0, stream>>>(buf_a, g_gamma, g_beta, g_mean, g_var, pooled);

    // 4) FC + top-1 routing + aux loss (last output element)
    route_kernel<<<1, 32, 0, stream>>>(pooled, g_fcw, g_fcb, route, out + (out_size - 1));

    // 5) routed expert conv1 (only the selected expert per sample)
    conv_wmma<float, __bf16, 7, 7><<<g1, 256, 0, stream>>>(
        x, wg_c1, nullptr, route, buf_a,
        15, 512, 512, 2, 3, 256, 256, 64, 64, 735, 768, 1);

    // 6) routed expert conv2
    dim3 g2(16384 / 64, 2, 8);
    conv_wmma<__bf16, __bf16, 3, 3><<<g2, 256, 0, stream>>>(
        buf_a, wg_c2, nullptr, route, buf_b,
        64, 256, 256, 2, 1, 128, 128, 128, 128, 576, 576, 1);

    // 7) routed head conv1 (+bias, relu)
    conv_wmma<__bf16, __bf16, 3, 3><<<g2, 256, 0, stream>>>(
        buf_b, wg_h1, e_h1b, route, buf_c,
        128, 128, 128, 1, 1, 128, 128, 128, 128, 1152, 1152, 1);

    // 8) routed head conv2 (1x1, +bias)
    dim3 g3(16384 / 64, 1, 8);
    conv_wmma<__bf16, float, 1, 1><<<g3, 256, 0, stream>>>(
        buf_c, wg_h2, e_h2b, route, buf_h2,
        128, 128, 128, 1, 0, 128, 128, 5, 16, 128, 128, 0);

    // 9) scatter into (hm | wh | reg); aux already written
    scatter_out<<<(8 * 5 * 16384 + 255)/256, 256, 0, stream>>>(buf_h2, out);
}
